// RadioPointTransformerV3_45286135169037
// MI455X (gfx1250) — matrix operand order repaired and checked by reading, not compiled, not run on hardware
//
#include <hip/hip_runtime.h>
#include <math.h>

// ---- problem constants (match reference) ----
#define N_PTS  524288
#define SEGS   64
#define C_DIM  512
#define H_DIM  256
#define LN_EPS 1e-5f

typedef __attribute__((ext_vector_type(16))) __bf16 v16bf;
typedef __attribute__((ext_vector_type(8)))  float  v8f;

// ============================================================================
// Kernel 0: repack w1 (fp32 [C,H]) -> bf16 in WMMA B-fragment order, zero out.
// Packed index: tid = ((kt*16 + nt)*32 + lane)*16 + j
//   n = nt*16 + (lane&15);  K = kt*32 + (lane>>4)*8 + (j&7) + (j>>3)*16
// so a wave can fetch one 32x16 B tile with two coalesced float4 loads/lane.
// ============================================================================
__global__ void prep_kernel(const float* __restrict__ w1,
                            unsigned short* __restrict__ packed,
                            float* __restrict__ out) {
    int tid  = blockIdx.x * blockDim.x + threadIdx.x;   // 131072 threads
    int j    = tid & 15;
    int lane = (tid >> 4) & 31;
    int nt   = (tid >> 9) & 15;
    int kt   = (tid >> 13) & 15;
    int n = nt * 16 + (lane & 15);
    int K = kt * 32 + ((lane >> 4) << 3) + (j & 7) + ((j >> 3) << 4);
    union { __bf16 h; unsigned short u; } cv;
    cv.h = (__bf16)w1[K * H_DIM + n];
    packed[tid] = cv.u;
    if (tid < SEGS * C_DIM) out[tid] = 0.0f;            // pooled accumulators
}

// ============================================================================
// Kernel 1: fused LayerNorm -> bf16 WMMA GEMM (x @ w1) -> +b1 -> exact GELU
//           -> dot(w2) => per-point score.  64 rows/block, 8 waves.
// ============================================================================
#define A_STRIDE_U 260   // 520 bf16 per row (512 + 8 pad) in uint units

__launch_bounds__(256)
__global__ void score_kernel(const float* __restrict__ feats,
                             const float* __restrict__ lnw,
                             const float* __restrict__ lnb,
                             const unsigned short* __restrict__ packedW1,
                             const float* __restrict__ b1,
                             const float* __restrict__ w2,
                             float* __restrict__ scores) {
    __shared__ __align__(16) unsigned sA[64 * A_STRIDE_U];  // 64x512 bf16 (padded)
    __shared__ float sS[64];

    const int tid  = threadIdx.x;
    const int lane = tid & 31;
    const int wave = tid >> 5;
    const int rowBase = blockIdx.x * 64;

    if (tid < 64) sS[tid] = 0.0f;

    // ---- Stage 1: LayerNorm, write bf16 A tiles to LDS (8 rows per wave) ----
    const float2* f2 = (const float2*)feats;
    for (int jr = 0; jr < 8; ++jr) {
        const int rl   = wave * 8 + jr;
        const long grow = (long)(rowBase + rl);
        float2 v[8];
        float sum = 0.0f, ssq = 0.0f;
        #pragma unroll
        for (int i = 0; i < 8; ++i) {
            v[i] = f2[grow * 256 + lane + 32 * i];
            sum += v[i].x + v[i].y;
            ssq += v[i].x * v[i].x + v[i].y * v[i].y;
        }
        #pragma unroll
        for (int m = 16; m >= 1; m >>= 1) {
            sum += __shfl_xor(sum, m, 32);
            ssq += __shfl_xor(ssq, m, 32);
        }
        const float mu   = sum * (1.0f / 512.0f);
        const float var  = ssq * (1.0f / 512.0f) - mu * mu;
        const float rstd = rsqrtf(var + LN_EPS);
        #pragma unroll
        for (int i = 0; i < 8; ++i) {
            const int c = 2 * (lane + 32 * i);
            const float x0 = (v[i].x - mu) * rstd * lnw[c]     + lnb[c];
            const float x1 = (v[i].y - mu) * rstd * lnw[c + 1] + lnb[c + 1];
            union { __bf16 h[2]; unsigned u; } pk;
            pk.h[0] = (__bf16)x0;
            pk.h[1] = (__bf16)x1;
            sA[rl * A_STRIDE_U + lane + 32 * i] = pk.u;
        }
    }
    __syncthreads();

    // ---- Stage 2: WMMA GEMM. wave -> (mtile = wave>>1, 8 ntiles) ----
    const int mtile = wave >> 1;
    const int nhalf = wave & 1;

    v8f acc[8] = {};  // 8 x 16x16 f32 accumulators (64 VGPRs)

    const int arow  = mtile * 16 + (lane & 15);
    const int kaoff = (lane >> 4) * 4;          // A K-interleave, uint units
    const float4* pw = (const float4*)packedW1;

    for (int kt = 0; kt < 16; ++kt) {
        union { v16bf v; float4 f[2]; } a;
        const unsigned* ap = &sA[arow * A_STRIDE_U + kt * 16 + kaoff];
        a.f[0] = *(const float4*)ap;            // K = kt*32 + kb .. +7
        a.f[1] = *(const float4*)(ap + 8);      // K = kt*32 + 16 + kb .. +7
        #pragma unroll
        for (int t = 0; t < 8; ++t) {
            const int ntile = nhalf * 8 + t;
            union { v16bf v; float4 f[2]; } b;
            const long bidx = ((long)(kt * 16 + ntile) * 32 + lane) * 2;
            b.f[0] = pw[bidx];
            b.f[1] = pw[bidx + 1];
            acc[t] = __builtin_amdgcn_wmma_f32_16x16x32_bf16(
                false, a.v, false, b.v, (short)0, acc[t], false, false);
        }
    }

    // ---- Stage 3: epilogue: +b1, exact GELU, * w2, reduce to scores ----
    float partial[8];
    #pragma unroll
    for (int r = 0; r < 8; ++r) partial[r] = 0.0f;

    #pragma unroll
    for (int t = 0; t < 8; ++t) {
        const int n = (nhalf * 8 + t) * 16 + (lane & 15);
        const float bb = b1[n];
        const float ww = w2[n];
        #pragma unroll
        for (int r = 0; r < 8; ++r) {
            const float x = acc[t][r] + bb;
            const float g = 0.5f * x * (1.0f + erff(x * 0.70710678118654752f));
            partial[r] += g * ww;
        }
    }
    // reduce across the 16 lanes holding the same rows
    #pragma unroll
    for (int m = 8; m >= 1; m >>= 1)
        #pragma unroll
        for (int r = 0; r < 8; ++r)
            partial[r] += __shfl_xor(partial[r], m, 32);

    if ((lane & 15) == 0) {
        const int mbase = mtile * 16 + ((lane >> 4) ? 8 : 0);
        #pragma unroll
        for (int r = 0; r < 8; ++r)
            atomicAdd(&sS[mbase + r], partial[r]);   // ds_add_f32, combines nhalf pair
    }
    __syncthreads();
    if (tid < 64) scores[rowBase + tid] = sS[tid];   // b2 omitted: softmax-invariant
}

// ============================================================================
// Kernel 2: per-segment max and sum(exp(s-m)); one block per segment.
// ============================================================================
__global__ void segstat_kernel(const float* __restrict__ scores,
                               const int* __restrict__ offs,
                               float* __restrict__ m_arr,
                               float* __restrict__ rz_arr) {
    __shared__ float red[256];
    const int b = blockIdx.x, tid = threadIdx.x;
    const int s0 = offs[b], s1 = offs[b + 1];

    float mx = -3.402823466e38f;
    for (int i = s0 + tid; i < s1; i += 256) mx = fmaxf(mx, scores[i]);
    red[tid] = mx; __syncthreads();
    for (int st = 128; st > 0; st >>= 1) {
        if (tid < st) red[tid] = fmaxf(red[tid], red[tid + st]);
        __syncthreads();
    }
    const float m = red[0];
    __syncthreads();

    float sum = 0.0f;
    for (int i = s0 + tid; i < s1; i += 256) sum += expf(scores[i] - m);
    red[tid] = sum; __syncthreads();
    for (int st = 128; st > 0; st >>= 1) {
        if (tid < st) red[tid] += red[tid + st];
        __syncthreads();
    }
    if (tid == 0) { m_arr[b] = m; rz_arr[b] = 1.0f / red[0]; }
}

// ============================================================================
// Kernel 3: pooled[b,c] += softmax_w(i) * feats[i,c].  256 rows per block;
// rows are segment-sorted, so flush atomics only on segment change.
// ============================================================================
__global__ void pool_kernel(const float* __restrict__ feats,
                            const float* __restrict__ scores,
                            const int* __restrict__ offs,
                            const float* __restrict__ m_arr,
                            const float* __restrict__ rz_arr,
                            float* __restrict__ out) {
    __shared__ float wsh[256];
    __shared__ int   ssh[256];
    __shared__ int   offsh[SEGS + 1];

    const int tid  = threadIdx.x;
    const int base = blockIdx.x * 256;
    if (tid <= SEGS) offsh[tid] = offs[tid];
    __syncthreads();

    // phase A: one row per thread -> segment id + softmax weight
    const int row = base + tid;
    int lo = 0, hi = SEGS - 1;
    while (lo < hi) {
        const int mid = (lo + hi + 1) >> 1;
        if (offsh[mid] <= row) lo = mid; else hi = mid - 1;
    }
    ssh[tid] = lo;
    wsh[tid] = expf(scores[row] - m_arr[lo]) * rz_arr[lo];
    __syncthreads();

    // phase B: two columns per thread across all 256 rows
    const float2* f2 = (const float2*)feats;
    float2 acc = {0.0f, 0.0f};
    int cur = ssh[0];
    for (int r = 0; r < 256; ++r) {
        const int sg = ssh[r];
        if (sg != cur) {
            atomicAdd(&out[cur * C_DIM + 2 * tid],     acc.x);
            atomicAdd(&out[cur * C_DIM + 2 * tid + 1], acc.y);
            acc.x = 0.0f; acc.y = 0.0f; cur = sg;
        }
        const float  w = wsh[r];
        const float2 v = f2[(long)(base + r) * 256 + tid];
        acc.x += w * v.x;
        acc.y += w * v.y;
    }
    atomicAdd(&out[cur * C_DIM + 2 * tid],     acc.x);
    atomicAdd(&out[cur * C_DIM + 2 * tid + 1], acc.y);
}

// ============================================================================
extern "C" void kernel_launch(void* const* d_in, const int* in_sizes, int n_in,
                              void* d_out, int out_size, void* d_ws, size_t ws_size,
                              hipStream_t stream) {
    const float* feats = (const float*)d_in[0];
    const float* lnw   = (const float*)d_in[1];
    const float* lnb   = (const float*)d_in[2];
    const float* w1    = (const float*)d_in[3];
    const float* b1    = (const float*)d_in[4];
    const float* w2    = (const float*)d_in[5];
    // d_in[6] = b2 : additive constant on scores -> softmax-invariant, unused
    const int*   offs  = (const int*)d_in[7];
    float* out = (float*)d_out;

    char* ws = (char*)d_ws;
    unsigned short* packed = (unsigned short*)ws;                 // 256 KB
    float* scores = (float*)(ws + 262144);                        // 2 MB
    float* m_arr  = (float*)(ws + 262144 + (size_t)N_PTS * 4);    // 256 B
    float* rz_arr = m_arr + SEGS;                                 // 256 B

    prep_kernel   <<<(C_DIM * H_DIM) / 256, 256, 0, stream>>>(w1, packed, out);
    score_kernel  <<<N_PTS / 64,            256, 0, stream>>>(feats, lnw, lnb,
                                                              packed, b1, w2, scores);
    segstat_kernel<<<SEGS,                  256, 0, stream>>>(scores, offs, m_arr, rz_arr);
    pool_kernel   <<<N_PTS / 256,           256, 0, stream>>>(feats, scores, offs,
                                                              m_arr, rz_arr, out);
}
